// Maploss_shwang_3358664425476
// MI455X (gfx1250) — compile-verified, matrix operand order checked by reading, not run on hardware
//
#include <hip/hip_runtime.h>
#include <hip/hip_bf16.h>

typedef __attribute__((ext_vector_type(2)))  float    v2f;
typedef __attribute__((ext_vector_type(8)))  float    v8f;
typedef __attribute__((ext_vector_type(16))) _Float16 v16h;

#define HIST 4096      // bins = top 12 bits of nonnegative float pattern
#define CHUNK 4096     // elements per block in pass1
#define T1 256         // pass1 block size (8 waves on wave32)

// ---------------------------------------------------------------------------
// Pass 1: fused loss compute + pos/neg stats + per-pair value histogram.
// grid = (chunks, pairs) where pair = t*B + b, t in {region, affinity}.
// ---------------------------------------------------------------------------
__global__ void maploss_pass1(const float* __restrict__ rlab,
                              const float* __restrict__ alab,
                              const float* __restrict__ rpre,
                              const float* __restrict__ apre,
                              const float* __restrict__ mask,
                              unsigned*    __restrict__ gcnt,
                              float*       __restrict__ gsum,
                              float*       __restrict__ P,   // [chunks_pad][cols]
                              int B, int N, int cols) {
  __shared__ unsigned hcnt[HIST];
  __shared__ float    hsum[HIST];
  __shared__ float    blk[4];

  const int pair  = blockIdx.y;
  const int t     = pair / B;
  const int b     = pair - t * B;
  const int chunk = blockIdx.x;
  const int tid   = threadIdx.x;

  for (int i = tid; i < HIST; i += T1) { hcnt[i] = 0u; hsum[i] = 0.0f; }
  if (tid < 4) blk[tid] = 0.0f;
  __syncthreads();

  const float* lab = (t == 0) ? rlab : alab;
  const float* pre = (t == 0) ? rpre : apre;
  const size_t sbase = (size_t)b * (size_t)N;

  float pos_sum = 0.f, npos = 0.f, neg_sum = 0.f, nneg = 0.f;

#pragma unroll
  for (int it = 0; it < CHUNK / (T1 * 4); ++it) {
    const int e0 = chunk * CHUNK + it * (T1 * 4) + tid * 4;
    if (e0 + 3 < N) {
      const size_t idx = sbase + (size_t)e0;
      // streaming prefetch of the next iteration's lines (global_prefetch_b8)
      __builtin_prefetch(lab  + idx + T1 * 4, 0, 0);
      __builtin_prefetch(pre  + idx + T1 * 4, 0, 0);
      __builtin_prefetch(mask + idx + T1 * 4, 0, 0);
      float4 l4 = *(const float4*)(lab  + idx);
      float4 p4 = *(const float4*)(pre  + idx);
      float4 m4 = *(const float4*)(mask + idx);
      float ls[4] = {l4.x, l4.y, l4.z, l4.w};
      float ps[4] = {p4.x, p4.y, p4.z, p4.w};
      float ms[4] = {m4.x, m4.y, m4.z, m4.w};
#pragma unroll
      for (int j = 0; j < 4; ++j) {
        const float d = ps[j] - ls[j];
        const float v = d * d * ms[j];           // >= 0, so float bits monotonic
        if (ls[j] >= 0.1f) { pos_sum += v; npos += 1.0f; }
        else {
          neg_sum += v; nneg += 1.0f;
          const unsigned bin = __float_as_uint(v) >> 19;   // 0..4095
          atomicAdd(&hcnt[bin], 1u);
          atomicAdd(&hsum[bin], v);
        }
      }
    } else {                                      // scalar tail (generic N)
      for (int j = 0; j < 4; ++j) {
        const int e = e0 + j;
        if (e < N) {
          const size_t idx = sbase + (size_t)e;
          const float l = lab[idx], p = pre[idx], m = mask[idx];
          const float d = p - l;
          const float v = d * d * m;
          if (l >= 0.1f) { pos_sum += v; npos += 1.0f; }
          else {
            neg_sum += v; nneg += 1.0f;
            const unsigned bin = __float_as_uint(v) >> 19;
            atomicAdd(&hcnt[bin], 1u);
            atomicAdd(&hsum[bin], v);
          }
        }
      }
    }
  }

  // wave32 reduction, then one LDS atomic per wave
#pragma unroll
  for (int off = 16; off > 0; off >>= 1) {
    pos_sum += __shfl_xor(pos_sum, off, 32);
    npos    += __shfl_xor(npos,    off, 32);
    neg_sum += __shfl_xor(neg_sum, off, 32);
    nneg    += __shfl_xor(nneg,    off, 32);
  }
  if ((tid & 31) == 0) {
    atomicAdd(&blk[0], npos);   atomicAdd(&blk[1], pos_sum);
    atomicAdd(&blk[2], nneg);   atomicAdd(&blk[3], neg_sum);
  }
  __syncthreads();

  for (int i = tid; i < HIST; i += T1) {
    const unsigned c = hcnt[i];
    if (c) {
      atomicAdd(&gcnt[(size_t)pair * HIST + i], c);
      atomicAdd(&gsum[(size_t)pair * HIST + i], hsum[i]);
    }
  }
  if (tid == 0) {
    float* row = P + (size_t)chunk * (size_t)cols + (size_t)pair * 4;
    row[0] = blk[0]; row[1] = blk[1]; row[2] = blk[2]; row[3] = blk[3];
  }
}

// ---------------------------------------------------------------------------
// Stage 2: WMMA column-sum. sums[col] = sum over rows of P[row][col].
// A = ones, so D[m][n] accumulates exact column sums; row->K mapping is
// irrelevant to the result (sum is permutation invariant). One wave per
// 16-column group; EXEC all-ones through the WMMA loop.
// ---------------------------------------------------------------------------
__global__ void maploss_colsum_wmma(const float* __restrict__ P,
                                    float* __restrict__ sums,
                                    int rows, int cols) {
  const int lane = threadIdx.x;                 // 0..31
  const int col  = blockIdx.x * 16 + (lane & 15);
  v8f acc = {};
#if defined(__HIP_DEVICE_COMPILE__) && __has_builtin(__builtin_amdgcn_wmma_f32_16x16x4_f32)
  v2f a; a.x = 1.0f; a.y = 1.0f;                // A = ones (16x4 f32, 2 VGPRs)
  for (int k = 0; k < rows; k += 4) {
    const int r0 = k + ((lane >> 4) << 1);      // lanes 0-15: k,k+1; 16-31: k+2,k+3
    v2f bm;
    bm.x = P[(size_t)r0 * (size_t)cols + col];
    bm.y = P[(size_t)(r0 + 1) * (size_t)cols + col];
    acc = __builtin_amdgcn_wmma_f32_16x16x4_f32(
        false, a, false, bm, (short)0, acc, false, false);
  }
#else
  v16h a;
#pragma unroll
  for (int j = 0; j < 16; ++j) a[j] = (_Float16)1.0f;
  for (int k = 0; k < rows; k += 32) {
    const int rbase = k + ((lane >> 4) << 4);   // lanes 0-15: k..; 16-31: k+16..
    v16h bm;
#pragma unroll
    for (int j = 0; j < 16; ++j)
      bm[j] = (_Float16)P[(size_t)(rbase + j) * (size_t)cols + col];
    acc = __builtin_amdgcn_wmma_f32_16x16x32_f16(
        false, a, false, bm, (short)0, acc, false, false);
  }
#endif
  if (lane < 16) sums[col] = acc[0];            // all D rows identical = colsum
}

// ---------------------------------------------------------------------------
// Stage 3: per-pair hard-negative selection via histogram suffix scan.
// ---------------------------------------------------------------------------
__global__ void maploss_select(const unsigned* __restrict__ gcnt,
                               const float*    __restrict__ gsum,
                               const float*    __restrict__ sums,
                               const int*      __restrict__ neg_rto_p,
                               const int*      __restrict__ nmin_p,
                               float*          __restrict__ result) {
  __shared__ unsigned hc[HIST];
  __shared__ float    hs[HIST];
  const int pair = blockIdx.x;
  for (int i = threadIdx.x; i < HIST; i += blockDim.x) {
    hc[i] = gcnt[(size_t)pair * HIST + i];
    hs[i] = gsum[(size_t)pair * HIST + i];
  }
  __syncthreads();
  if (threadIdx.x == 0) {
    const float npos    = sums[pair * 4 + 0];
    const float pos_sum = sums[pair * 4 + 1];
    const float nneg    = sums[pair * 4 + 2];
    const float neg_sum = sums[pair * 4 + 3];
    const float neg_rto = (float)(*neg_rto_p);
    const float nmin    = (float)(*nmin_p);

    const float n_hard = fmaxf(nmin, neg_rto * npos);
    const float k_hard = floorf(n_hard);        // torch int(n_hard)
    const float k_min  = nmin;

    float cumc = 0.f, cums = 0.f;
    float topk_hard = 0.f, topk_min = 0.f;
    bool done_h = false, done_m = false;
    for (int i = HIST - 1; i >= 0; --i) {       // descending values
      const float c = (float)hc[i];
      const float s = hs[i];
      if (!done_h && c > 0.f && cumc + c >= k_hard) {
        topk_hard = cums + (k_hard - cumc) * (s / c);
        done_h = true;
      }
      if (!done_m && c > 0.f && cumc + c >= k_min) {
        topk_min = cums + (k_min - cumc) * (s / c);
        done_m = true;
      }
      cumc += c; cums += s;
    }
    if (!done_h) topk_hard = cums;              // k beyond negatives: zeros pad
    if (!done_m) topk_min  = cums;

    const float pos_loss = pos_sum / fmaxf(npos, 1e-12f);
    float neg_loss;
    if (npos == 0.f)                 neg_loss = topk_min / nmin;
    else if (nneg < neg_rto * npos)  neg_loss = neg_sum / nneg;
    else                             neg_loss = topk_hard / n_hard;
    result[pair] = pos_loss + neg_loss;
  }
}

// ---------------------------------------------------------------------------
// Stage 4: one-wave final reduction to the scalar output.
// ---------------------------------------------------------------------------
__global__ void maploss_finalize(const float* __restrict__ result,
                                 float* __restrict__ out, int n, float invB) {
  float s = 0.f;
  for (int i = threadIdx.x; i < n; i += 32) s += result[i];
#pragma unroll
  for (int off = 16; off > 0; off >>= 1) s += __shfl_xor(s, off, 32);
  if (threadIdx.x == 0) out[0] = s * invB;
}

// ---------------------------------------------------------------------------
extern "C" void kernel_launch(void* const* d_in, const int* in_sizes, int n_in,
                              void* d_out, int out_size, void* d_ws, size_t ws_size,
                              hipStream_t stream) {
  const float* rlab = (const float*)d_in[0];
  const float* alab = (const float*)d_in[1];
  const float* rpre = (const float*)d_in[2];
  const float* apre = (const float*)d_in[3];
  const float* mask = (const float*)d_in[4];
  const int* neg_rto_p = (const int*)d_in[5];
  const int* nmin_p    = (const int*)d_in[6];

  const int B     = 32;
  const int N     = in_sizes[0] / B;            // H*W per sample
  const int pairs = 2 * B;                      // 64
  const int cols  = pairs * 4;                  // 256 partial-scalar columns
  const int chunks = (N + CHUNK - 1) / CHUNK;
  const int chunks_pad = ((chunks + 31) / 32) * 32;  // rows multiple of 32

  // workspace carve-out
  size_t off = 0;
  unsigned* gcnt = (unsigned*)((char*)d_ws + off); off += (size_t)pairs * HIST * sizeof(unsigned);
  float* gsum    = (float*)((char*)d_ws + off);    off += (size_t)pairs * HIST * sizeof(float);
  float* P       = (float*)((char*)d_ws + off);    off += (size_t)chunks_pad * cols * sizeof(float);
  float* sums    = (float*)((char*)d_ws + off);    off += (size_t)cols * sizeof(float);
  float* result  = (float*)((char*)d_ws + off);    off += (size_t)pairs * sizeof(float);

  hipMemsetAsync(d_ws, 0, off, stream);         // async: graph-capture safe

  dim3 g1(chunks, pairs);
  maploss_pass1<<<g1, T1, 0, stream>>>(rlab, alab, rpre, apre, mask,
                                       gcnt, gsum, P, B, N, cols);
  maploss_colsum_wmma<<<cols / 16, 32, 0, stream>>>(P, sums, chunks_pad, cols);
  maploss_select<<<pairs, 256, 0, stream>>>(gcnt, gsum, sums, neg_rto_p, nmin_p, result);
  maploss_finalize<<<1, 32, 0, stream>>>(result, (float*)d_out, pairs, 1.0f / (float)B);
}
